// StateFormer_1675037246123
// MI455X (gfx1250) — matrix operand order repaired and checked by reading, not compiled, not run on hardware
//
#include <hip/hip_runtime.h>
#include <hip/hip_bf16.h>
#include <math.h>

#define NN 32768
#define EE 524288
#define DD 128
#define DI 256      // D_INNER
#define DS 16       // D_STATE
#define DTR 8       // DT_RANK
#define CLEN 128
#define NCHUNK 256  // NN / CLEN
#define NCH 4096    // DI * DS

typedef __attribute__((ext_vector_type(16))) __bf16 v16bf;
typedef __attribute__((ext_vector_type(8)))  __bf16 v8bf;
typedef __attribute__((ext_vector_type(8)))  float  v8f;

__device__ __forceinline__ float sigmoidf_(float x){ return 1.0f/(1.0f+__expf(-x)); }
__device__ __forceinline__ float siluf_(float x){ return x*sigmoidf_(x); }
__device__ __forceinline__ float softplusf_(float x){ return (x>20.0f)?x:log1pf(__expf(x)); }

#define WMMA_BF16(a,b,c) __builtin_amdgcn_wmma_f32_16x16x32_bf16(false,(a),false,(b),(short)0,(c),false,false)
#define CAT16(lo,hi) __builtin_shufflevector((lo),(hi),0,1,2,3,4,5,6,7,8,9,10,11,12,13,14,15)

// A operand from pre-converted bf16 row-major X: two b128 loads, no cvt.
// Lanes 0-15: row M=lane&15, K={0..7,16..23}; lanes 16-31: K={8..15,24..31}.
__device__ __forceinline__ v16bf ld_a_bf(const __bf16* __restrict__ X, int ld, int m0, int k0, int lane){
  int m = m0 + (lane & 15);
  const __bf16* p = X + (size_t)m*ld + k0 + ((lane>>4)<<3);
  v8bf lo = *(const v8bf*)p;
  v8bf hi = *(const v8bf*)(p+16);
  return CAT16(lo, hi);
}
// B operand from pre-converted bf16 row-major W (Bmat[k][n]=W[n][k]): lanes 0-15 K={0..7,16..23}; lanes 16-31 K={8..15,24..31}.
__device__ __forceinline__ v16bf ld_b_bf(const __bf16* __restrict__ W, int ld, int n0, int k0, int lane){
  int n = n0 + (lane & 15);
  const __bf16* p = W + (size_t)n*ld + k0 + ((lane>>4)<<3);
  v8bf lo = *(const v8bf*)p;
  v8bf hi = *(const v8bf*)(p+16);
  return CAT16(lo, hi);
}

// Async global->LDS staging of one b128 (16B) per call (ASYNCcnt-tracked).
__device__ __forceinline__ void async_b128(uint32_t lds_off, const void* gptr){
  asm volatile("global_load_async_to_lds_b128 %0, %1, off"
               :: "v"(lds_off), "v"((unsigned long long)(uintptr_t)gptr) : "memory");
}
__device__ __forceinline__ void wait_async0(){
  asm volatile("s_wait_asynccnt 0" ::: "memory");
}
__device__ __forceinline__ uint32_t lds_off32(const void* p){
  return (uint32_t)(uintptr_t)p;   // low 32 bits of shared-aperture flat address = LDS byte offset
}

// ---------------- f32 -> bf16 conversion ----------------
__global__ void k_cvt(const float* __restrict__ in, __bf16* __restrict__ outp, int n){
  int g = blockIdx.x*blockDim.x + threadIdx.x;
  if (g < n) outp[g] = (__bf16)in[g];
}

// ---------------- Edge MLP: BN -> silu(h@W1^T+b1) -> sigmoid(sigmoid(h@W2^T+b2)) ----------------
__global__ void k_edge_mlp(const float* __restrict__ attr,
                           const float* __restrict__ g1, const float* __restrict__ be1,
                           const float* __restrict__ mu1, const float* __restrict__ var1,
                           const __bf16* __restrict__ w1bf, const float* __restrict__ b1,
                           const float* __restrict__ w2, const float* __restrict__ b2,
                           float* __restrict__ ew)
{
  __shared__ float sc[DD], sb[DD];
  __shared__ float ht[8][16][32];
  int tid = threadIdx.x, wave = tid>>5, lane = tid&31;
  if (tid < DD){
    float s = g1[tid]*rsqrtf(var1[tid]+1e-5f);
    sc[tid] = s; sb[tid] = be1[tid]-mu1[tid]*s;
  }
  __syncthreads();
  int e0 = blockIdx.x*128 + wave*16;
  int m  = e0 + (lane&15);
  int koffA = (lane>>4)<<3;
  v8f c0 = {}, c1 = {};
#pragma unroll
  for (int k0=0;k0<DD;k0+=32){
    v16bf a;
#pragma unroll
    for (int j=0;j<8;++j){ int k=k0+koffA+j;    a[j]  =(__bf16)(attr[(size_t)m*DD+k]*sc[k]+sb[k]); }
#pragma unroll
    for (int j=0;j<8;++j){ int k=k0+koffA+16+j; a[8+j]=(__bf16)(attr[(size_t)m*DD+k]*sc[k]+sb[k]); }
    v16bf bb0 = ld_b_bf(w1bf, DD, 0,  k0, lane);
    v16bf bb1 = ld_b_bf(w1bf, DD, 16, k0, lane);
    c0 = WMMA_BF16(a, bb0, c0);
    c1 = WMMA_BF16(a, bb1, c1);
  }
  int n = lane&15, mh = (lane>>4)<<3;
#pragma unroll
  for (int i=0;i<8;++i){
    ht[wave][i+mh][n]    = siluf_(c0[i]+b1[n]);
    ht[wave][i+mh][n+16] = siluf_(c1[i]+b1[n+16]);
  }
  __syncthreads();
  if (lane < 16){
    float acc = b2[0];
#pragma unroll
    for (int j=0;j<32;++j) acc += ht[wave][lane][j]*w2[j];
    ew[e0+lane] = sigmoidf_(sigmoidf_(acc));
  }
}

// ---------------- in_proj: xz = x @ in_proj_w^T  (16x64 outputs per wave) ----------------
__global__ void k_inproj(const __bf16* __restrict__ xbf, const __bf16* __restrict__ wbf,
                         float* __restrict__ out)
{
  int tid=threadIdx.x, wave=tid>>5, lane=tid&31;
  int m0 = blockIdx.x*16;
  int n0 = wave*64;
  v8f c0={},c1={},c2={},c3={};
#pragma unroll
  for (int k0=0;k0<DD;k0+=32){
    v16bf a = ld_a_bf(xbf, DD, m0, k0, lane);
    c0 = WMMA_BF16(a, ld_b_bf(wbf, DD, n0,    k0, lane), c0);
    c1 = WMMA_BF16(a, ld_b_bf(wbf, DD, n0+16, k0, lane), c1);
    c2 = WMMA_BF16(a, ld_b_bf(wbf, DD, n0+32, k0, lane), c2);
    c3 = WMMA_BF16(a, ld_b_bf(wbf, DD, n0+48, k0, lane), c3);
  }
  int n = n0+(lane&15), mh=(lane>>4)<<3;
#pragma unroll
  for (int i=0;i<8;++i){
    size_t r = (size_t)(m0+i+mh)*(2*DI);
    out[r+n]    = c0[i];
    out[r+n+16] = c1[i];
    out[r+n+32] = c2[i];
    out[r+n+48] = c3[i];
  }
}

// ---------------- depthwise causal conv (k=4) + silu -> u (f32 + bf16 mirror) ----------------
__global__ void k_conv(const float* __restrict__ xz, const float* __restrict__ cw,
                       const float* __restrict__ cb, float* __restrict__ u,
                       __bf16* __restrict__ ubf)
{
  int g = blockIdx.x*blockDim.x + threadIdx.x;
  int d = g & (DI-1), t = g >> 8;
  float acc = cb[d];
  if (t>=3) acc += cw[d*4+0]*xz[(size_t)(t-3)*(2*DI)+d];
  if (t>=2) acc += cw[d*4+1]*xz[(size_t)(t-2)*(2*DI)+d];
  if (t>=1) acc += cw[d*4+2]*xz[(size_t)(t-1)*(2*DI)+d];
  acc += cw[d*4+3]*xz[(size_t)t*(2*DI)+d];
  float v = siluf_(acc);
  u[(size_t)t*DI+d] = v;
  ubf[(size_t)t*DI+d] = (__bf16)v;
}

// ---------------- Wcomb = dt_proj_w (256x8) @ x_proj_w[0:8] (8x256), stored bf16 ----------------
__global__ void k_wcomb(const float* __restrict__ dtw, const float* __restrict__ xpw,
                        __bf16* __restrict__ wcbf)
{
  int d = blockIdx.x, k = threadIdx.x;
  float acc = 0.f;
#pragma unroll
  for (int r=0;r<DTR;++r) acc += dtw[d*DTR+r]*xpw[r*DI+k];
  wcbf[d*DI+k] = (__bf16)acc;
}

// ---------------- delta = softplus(u @ Wcomb^T + dt_bias)  (16x32 per wave) ----------------
__global__ void k_delta(const __bf16* __restrict__ ubf, const __bf16* __restrict__ wcbf,
                        const float* __restrict__ dtb, float* __restrict__ delta)
{
  int tid=threadIdx.x, wave=tid>>5, lane=tid&31;
  int m0 = blockIdx.x*16;
  int n0 = wave*32;
  v8f c0={},c1={};
#pragma unroll
  for (int k0=0;k0<DI;k0+=32){
    v16bf a = ld_a_bf(ubf, DI, m0, k0, lane);
    c0 = WMMA_BF16(a, ld_b_bf(wcbf, DI, n0,    k0, lane), c0);
    c1 = WMMA_BF16(a, ld_b_bf(wcbf, DI, n0+16, k0, lane), c1);
  }
  int n = n0+(lane&15), mh=(lane>>4)<<3;
  float bias0 = dtb[n], bias1 = dtb[n+16];
#pragma unroll
  for (int i=0;i<8;++i){
    size_t r = (size_t)(m0+i+mh)*DI;
    delta[r+n]    = softplusf_(c0[i]+bias0);
    delta[r+n+16] = softplusf_(c1[i]+bias1);
  }
}

// ---------------- B,C = u @ x_proj_w[8:40]^T ----------------
__global__ void k_bc(const __bf16* __restrict__ ubf, const __bf16* __restrict__ xpwbc,
                     float* __restrict__ Braw, float* __restrict__ Craw)
{
  int tid=threadIdx.x, wave=tid>>5, lane=tid&31;
  int m0 = blockIdx.x*16;
  v8f c = {};
#pragma unroll
  for (int k0=0;k0<DI;k0+=32){
    v16bf a = ld_a_bf(ubf, DI, m0, k0, lane);
    v16bf b = ld_b_bf(xpwbc, DI, wave*16, k0, lane);
    c = WMMA_BF16(a,b,c);
  }
  float* outp = wave ? Craw : Braw;
  int n = lane&15, mh=(lane>>4)<<3;
#pragma unroll
  for (int i=0;i<8;++i) outp[(size_t)(m0+i+mh)*DS+n] = c[i];
}

// ---------------- edge scatter: agg[dst] += ew * {B,C}[src] ----------------
__global__ void k_scatter(const int* __restrict__ ei, const float* __restrict__ ew,
                          const float* __restrict__ Braw, const float* __restrict__ Craw,
                          float* __restrict__ Bagg, float* __restrict__ Cagg)
{
  int g = blockIdx.x*blockDim.x + threadIdx.x;
  int e = g>>4, s = g&15;
  int src = ei[e], dst = ei[EE+e];
  float w = ew[e];
  atomicAdd(&Bagg[(size_t)dst*DS+s], w*Braw[(size_t)src*DS+s]);
  atomicAdd(&Cagg[(size_t)dst*DS+s], w*Craw[(size_t)src*DS+s]);
}

// ---------------- scan pass 1: per-chunk (P = prod dA, S = local state) ----------------
// B rows for the chunk are staged to LDS once per block via async-to-LDS (shared by all 16 d's).
__global__ void k_scan1(const float* __restrict__ delta, const float* __restrict__ u,
                        const float* __restrict__ Braw, const float* __restrict__ Bagg,
                        const float* __restrict__ alog,
                        float* __restrict__ cP, float* __restrict__ cS)
{
  __shared__ float sB[CLEN*DS];
  __shared__ float sBa[CLEN*DS];
  int tid=threadIdx.x;
  int dl = tid>>4, s = tid&15;
  int d = blockIdx.y*16 + dl;
  int c = blockIdx.x;
  // stage chunk B rows: 2048 floats per array = 512 b128 ops -> 2 per thread per array
  {
    const float* gB  = Braw + (size_t)c*CLEN*DS;
    const float* gBa = Bagg + (size_t)c*CLEN*DS;
    uint32_t lB  = lds_off32(sB);
    uint32_t lBa = lds_off32(sBa);
#pragma unroll
    for (int i=0;i<2;++i){
      int idx = tid + i*256;                 // b128 index
      async_b128(lB  + idx*16, gB  + idx*4);
      async_b128(lBa + idx*16, gBa + idx*4);
    }
  }
  float A = -__expf(alog[d*DS+s]);
  float P = 1.f, S = 0.f;
  int t0 = c*CLEN;
  wait_async0();
  __syncthreads();
  for (int tt=0; tt<CLEN; ++tt){
    int t = t0 + tt;
    __builtin_prefetch(&delta[(size_t)(t+16)*DI+d], 0, 1);
    __builtin_prefetch(&u[(size_t)(t+16)*DI+d], 0, 1);
    float dlt = delta[(size_t)t*DI+d];
    float uu  = u[(size_t)t*DI+d];
    float Bt  = sB[tt*DS+s] + sBa[tt*DS+s];
    float dA  = __expf(dlt*A);
    S = dA*S + dlt*Bt*uu;
    P *= dA;
  }
  int ch = d*DS+s;
  cP[(size_t)c*NCH+ch] = P;
  cS[(size_t)c*NCH+ch] = S;
}

// ---------------- scan pass 2: serial combine of chunk summaries ----------------
__global__ void k_scan2(const float* __restrict__ cP, const float* __restrict__ cS,
                        float* __restrict__ hinit)
{
  int ch = blockIdx.x*blockDim.x + threadIdx.x;
  float h = 0.f;
  hinit[ch] = 0.f;
  for (int c=0; c<NCHUNK-1; ++c){
    h = cP[(size_t)c*NCH+ch]*h + cS[(size_t)c*NCH+ch];
    hinit[(size_t)(c+1)*NCH+ch] = h;
  }
}

// ---------------- scan pass 3: replay with initial state, y_t = h_t . C_t ----------------
__global__ void k_scan3(const float* __restrict__ delta, const float* __restrict__ u,
                        const float* __restrict__ Braw, const float* __restrict__ Bagg,
                        const float* __restrict__ Craw, const float* __restrict__ Cagg,
                        const float* __restrict__ alog, const float* __restrict__ hinit,
                        float* __restrict__ ys)
{
  __shared__ float sB[CLEN*DS];
  __shared__ float sBa[CLEN*DS];
  __shared__ float sC[CLEN*DS];
  __shared__ float sCa[CLEN*DS];
  int tid=threadIdx.x;
  int dl = tid>>4, s = tid&15;
  int d = blockIdx.y*16 + dl;
  int c = blockIdx.x;
  {
    size_t base = (size_t)c*CLEN*DS;
    uint32_t lB  = lds_off32(sB),  lBa = lds_off32(sBa);
    uint32_t lC  = lds_off32(sC),  lCa = lds_off32(sCa);
#pragma unroll
    for (int i=0;i<2;++i){
      int idx = tid + i*256;
      async_b128(lB  + idx*16, Braw + base + idx*4);
      async_b128(lBa + idx*16, Bagg + base + idx*4);
      async_b128(lC  + idx*16, Craw + base + idx*4);
      async_b128(lCa + idx*16, Cagg + base + idx*4);
    }
  }
  int ch = d*DS+s;
  float A = -__expf(alog[d*DS+s]);
  float h = hinit[(size_t)c*NCH+ch];
  int t0 = c*CLEN;
  wait_async0();
  __syncthreads();
  for (int tt=0; tt<CLEN; ++tt){
    int t = t0 + tt;
    __builtin_prefetch(&delta[(size_t)(t+16)*DI+d], 0, 1);
    __builtin_prefetch(&u[(size_t)(t+16)*DI+d], 0, 1);
    float dlt = delta[(size_t)t*DI+d];
    float uu  = u[(size_t)t*DI+d];
    float Bt  = sB[tt*DS+s] + sBa[tt*DS+s];
    float Ct  = sC[tt*DS+s] + sCa[tt*DS+s];
    float dA  = __expf(dlt*A);
    h = dA*h + dlt*Bt*uu;
    float v = h*Ct;                 // 16-wide reduction over s within half-wave
    v += __shfl_xor(v, 1, 16);
    v += __shfl_xor(v, 2, 16);
    v += __shfl_xor(v, 4, 16);
    v += __shfl_xor(v, 8, 16);
    if (s==0) ys[(size_t)t*DI+d] = v;
  }
}

// ---------------- yf = (ys + u*D) * silu(z), stored bf16 for the final GEMM ----------------
__global__ void k_yf(const float* __restrict__ ys, const float* __restrict__ u,
                     const float* __restrict__ Dp, const float* __restrict__ xz,
                     __bf16* __restrict__ yfbf)
{
  int g = blockIdx.x*blockDim.x + threadIdx.x;
  int d = g & (DI-1); size_t t = (size_t)(g >> 8);
  float z = xz[t*(2*DI)+DI+d];
  yfbf[t*DI+d] = (__bf16)((ys[t*DI+d] + u[t*DI+d]*Dp[d]) * siluf_(z));
}

// ---------------- out = x + BN2(yf @ out_proj_w^T) ----------------
__global__ void k_out(const __bf16* __restrict__ yfbf, const __bf16* __restrict__ wbf,
                      const float* __restrict__ x,
                      const float* __restrict__ g2, const float* __restrict__ be2,
                      const float* __restrict__ mu2, const float* __restrict__ var2,
                      float* __restrict__ out)
{
  int tid=threadIdx.x, wave=tid>>5, lane=tid&31;
  int m0 = blockIdx.x*16;
  int n0 = wave*16;
  v8f c = {};
#pragma unroll
  for (int k0=0;k0<DI;k0+=32){
    v16bf a = ld_a_bf(yfbf, DI, m0, k0, lane);
    v16bf b = ld_b_bf(wbf, DI, n0, k0, lane);
    c = WMMA_BF16(a,b,c);
  }
  int n = n0+(lane&15), mh=(lane>>4)<<3;
  float sc = g2[n]*rsqrtf(var2[n]+1e-5f);
  float bs = be2[n]-mu2[n]*sc;
#pragma unroll
  for (int i=0;i<8;++i){
    size_t idx = (size_t)(m0+i+mh)*DD + n;
    out[idx] = x[idx] + c[i]*sc + bs;
  }
}

extern "C" void kernel_launch(void* const* d_in, const int* in_sizes, int n_in,
                              void* d_out, int out_size, void* d_ws, size_t ws_size,
                              hipStream_t stream) {
  const float* x     = (const float*)d_in[0];
  const int*   eidx  = (const int*)  d_in[1];
  const float* eattr = (const float*)d_in[2];
  const float* bn1g  = (const float*)d_in[3];
  const float* bn1b  = (const float*)d_in[4];
  const float* bn1m  = (const float*)d_in[5];
  const float* bn1v  = (const float*)d_in[6];
  const float* w1    = (const float*)d_in[7];
  const float* b1    = (const float*)d_in[8];
  const float* w2    = (const float*)d_in[9];
  const float* b2    = (const float*)d_in[10];
  const float* inpw  = (const float*)d_in[11];
  const float* convw = (const float*)d_in[12];
  const float* convb = (const float*)d_in[13];
  const float* xpw   = (const float*)d_in[14];
  const float* dtw   = (const float*)d_in[15];
  const float* dtb   = (const float*)d_in[16];
  const float* alog  = (const float*)d_in[17];
  const float* Dp    = (const float*)d_in[18];
  const float* outw  = (const float*)d_in[19];
  const float* bn2g  = (const float*)d_in[20];
  const float* bn2b  = (const float*)d_in[21];
  const float* bn2m  = (const float*)d_in[22];
  const float* bn2v  = (const float*)d_in[23];
  float* outp = (float*)d_out;

  char*  wsb = (char*)d_ws;
  size_t off = 0;
  auto alloc = [&](size_t bytes)->char*{ char* p = wsb + off; off += (bytes + 255) & ~(size_t)255; return p; };

  float* xz    = (float*)alloc((size_t)NN*2*DI*4);
  float* u     = (float*)alloc((size_t)NN*DI*4);
  float* dl    = (float*)alloc((size_t)NN*DI*4);
  float* ys    = (float*)alloc((size_t)NN*DI*4);
  float* Braw  = (float*)alloc((size_t)NN*DS*4);
  float* Craw  = (float*)alloc((size_t)NN*DS*4);
  float* Bagg  = (float*)alloc((size_t)NN*DS*4);
  float* Cagg  = (float*)alloc((size_t)NN*DS*4);
  float* ewb   = (float*)alloc((size_t)EE*4);
  float* cP    = (float*)alloc((size_t)NCHUNK*NCH*4);
  float* cS    = (float*)alloc((size_t)NCHUNK*NCH*4);
  float* hinit = (float*)alloc((size_t)NCHUNK*NCH*4);
  __bf16* xbf    = (__bf16*)alloc((size_t)NN*DD*2);
  __bf16* ubf    = (__bf16*)alloc((size_t)NN*DI*2);
  __bf16* yfbf   = (__bf16*)alloc((size_t)NN*DI*2);
  __bf16* inpwbf = (__bf16*)alloc((size_t)2*DI*DD*2);
  __bf16* wcbf   = (__bf16*)alloc((size_t)DI*DI*2);
  __bf16* xpwbcbf= (__bf16*)alloc((size_t)2*DS*DI*2);
  __bf16* outwbf = (__bf16*)alloc((size_t)DD*DI*2);
  __bf16* w1bf   = (__bf16*)alloc((size_t)32*DD*2);

  // weight / activation bf16 mirrors
  k_cvt<<<(NN*DD+255)/256, 256, 0, stream>>>(x, xbf, NN*DD);
  k_cvt<<<(2*DI*DD+255)/256, 256, 0, stream>>>(inpw, inpwbf, 2*DI*DD);
  k_cvt<<<(2*DS*DI+255)/256, 256, 0, stream>>>(xpw + DTR*DI, xpwbcbf, 2*DS*DI);
  k_cvt<<<(DD*DI+255)/256, 256, 0, stream>>>(outw, outwbf, DD*DI);
  k_cvt<<<(32*DD+255)/256, 256, 0, stream>>>(w1, w1bf, 32*DD);

  k_edge_mlp<<<EE/128, 256, 0, stream>>>(eattr, bn1g, bn1b, bn1m, bn1v, w1bf, b1, w2, b2, ewb);
  k_inproj<<<NN/16, 256, 0, stream>>>(xbf, inpwbf, xz);
  k_conv<<<(NN*DI)/256, 256, 0, stream>>>(xz, convw, convb, u, ubf);
  k_wcomb<<<DI, DI, 0, stream>>>(dtw, xpw, wcbf);
  k_delta<<<NN/16, 256, 0, stream>>>(ubf, wcbf, dtb, dl);
  k_bc<<<NN/16, 64, 0, stream>>>(ubf, xpwbcbf, Braw, Craw);
  hipMemsetAsync(Bagg, 0, (size_t)2*NN*DS*sizeof(float), stream);   // Bagg+Cagg contiguous
  k_scatter<<<(EE*16)/256, 256, 0, stream>>>(eidx, ewb, Braw, Craw, Bagg, Cagg);
  k_scan1<<<dim3(NCHUNK, DI/16), 256, 0, stream>>>(dl, u, Braw, Bagg, alog, cP, cS);
  k_scan2<<<NCH/256, 256, 0, stream>>>(cP, cS, hinit);
  k_scan3<<<dim3(NCHUNK, DI/16), 256, 0, stream>>>(dl, u, Braw, Bagg, Craw, Cagg, alog, hinit, ys);
  k_yf<<<(NN*DI)/256, 256, 0, stream>>>(ys, u, Dp, xz, yfbf);
  k_out<<<NN/16, 256, 0, stream>>>(yfbf, outwbf, x, bn2g, bn2b, bn2m, bn2v, outp);
}
